// NeighConv_33663953666895
// MI455X (gfx1250) — compile-verified, matrix operand order checked by reading, not run on hardware
//
#include <hip/hip_runtime.h>

typedef __attribute__((ext_vector_type(16))) _Float16 v16h;
typedef __attribute__((ext_vector_type(8)))  _Float16 v8h;
typedef __attribute__((ext_vector_type(8)))  float    v8f;

#define N_NODES 50000
#define KNEI    16
#define DFEAT   128
#define DOUT    128
#define DW      (2 * DFEAT)   // 256
#define WPB     8             // waves per block

// One wave32 owns 16 consecutive nodes.
// Phase 1: c[node][o] = ctr_node . W2^T for all 16 nodes in ONE 16x128x128 WMMA
//          pass (A rows = 16 ctr rows), parked in a per-wave LDS slab.
// Phase 2: per node, neighbor-only WMMA (16x128x128) + cosine weights computed
//          from the same gathered loads; epilogue adds c-row + bias, scales,
//          max-reduces over the 16 neighbors.
__global__ __launch_bounds__(256)
void neighconv_wmma(const float* __restrict__ feat,
                    const long long* __restrict__ nidx,
                    const float* __restrict__ W,
                    const float* __restrict__ bias,
                    float* __restrict__ out)
{
    __shared__ _Float16 sW[DOUT * DW];        // 128 x 256 f16 = 64 KB
    __shared__ float    sC[WPB][KNEI * DOUT]; // per-wave c blocks: 8 x 8 KB

    // ---- stage W into LDS as f16 (once per block) ----
    for (int i = threadIdx.x * 4; i < DOUT * DW; i += blockDim.x * 4) {
        float4 w4 = *(const float4*)(W + i);
        sW[i + 0] = (_Float16)w4.x;
        sW[i + 1] = (_Float16)w4.y;
        sW[i + 2] = (_Float16)w4.z;
        sW[i + 3] = (_Float16)w4.w;
    }
    __syncthreads();

    const int lane = threadIdx.x & 31;
    const int half = lane >> 4;     // K-subgroup select (ISA 16-bit A/B layout)
    const int mrow = lane & 15;     // matrix row (A) / column (B,C,D)
    const int kb   = half * 8;

    float bv[8];
#pragma unroll
    for (int t = 0; t < 8; ++t) bv[t] = bias[t * 16 + mrow];

    // B-tile: B[k][col] = W[t*16+col][k] (mirrored A layout, 2x ds_load_b128)
    auto loadB = [&](int t, int kw) -> v16h {
        const _Float16* wr = sW + (t * 16 + mrow) * DW + kw;
        v8h lo = *(const v8h*)(wr + kb);
        v8h hi = *(const v8h*)(wr + 16 + kb);
        v16h bm;
#pragma unroll
        for (int j = 0; j < 8; ++j) { bm[j] = lo[j]; bm[8 + j] = hi[j]; }
        return bm;
    };

    // A-tile from a row-major 128-f32 row; accumulates sum(row^2) on the side
    auto buildA = [&](const float* row, int k0, float& ss) -> v16h {
        float4 p0 = *(const float4*)(row + k0 + kb);
        float4 p1 = *(const float4*)(row + k0 + kb + 4);
        float4 q0 = *(const float4*)(row + k0 + 16 + kb);
        float4 q1 = *(const float4*)(row + k0 + 16 + kb + 4);
        ss += p0.x*p0.x + p0.y*p0.y + p0.z*p0.z + p0.w*p0.w;
        ss += p1.x*p1.x + p1.y*p1.y + p1.z*p1.z + p1.w*p1.w;
        ss += q0.x*q0.x + q0.y*q0.y + q0.z*q0.z + q0.w*q0.w;
        ss += q1.x*q1.x + q1.y*q1.y + q1.z*q1.z + q1.w*q1.w;
        v16h a;
        a[0]=(_Float16)p0.x;  a[1]=(_Float16)p0.y;  a[2]=(_Float16)p0.z;  a[3]=(_Float16)p0.w;
        a[4]=(_Float16)p1.x;  a[5]=(_Float16)p1.y;  a[6]=(_Float16)p1.z;  a[7]=(_Float16)p1.w;
        a[8]=(_Float16)q0.x;  a[9]=(_Float16)q0.y;  a[10]=(_Float16)q0.z; a[11]=(_Float16)q0.w;
        a[12]=(_Float16)q1.x; a[13]=(_Float16)q1.y; a[14]=(_Float16)q1.z; a[15]=(_Float16)q1.w;
        return a;
    };

    // Neighbor A-tile: also dots against ctr at the same K positions (L0 hits)
    auto buildA_dot = [&](const float* row, const float* cr, int k0,
                          float& dp, float& ss) -> v16h {
        float4 p0 = *(const float4*)(row + k0 + kb);
        float4 p1 = *(const float4*)(row + k0 + kb + 4);
        float4 q0 = *(const float4*)(row + k0 + 16 + kb);
        float4 q1 = *(const float4*)(row + k0 + 16 + kb + 4);
        float4 c0 = *(const float4*)(cr + k0 + kb);
        float4 c1 = *(const float4*)(cr + k0 + kb + 4);
        float4 d0 = *(const float4*)(cr + k0 + 16 + kb);
        float4 d1 = *(const float4*)(cr + k0 + 16 + kb + 4);
        dp += p0.x*c0.x + p0.y*c0.y + p0.z*c0.z + p0.w*c0.w;
        dp += p1.x*c1.x + p1.y*c1.y + p1.z*c1.z + p1.w*c1.w;
        dp += q0.x*d0.x + q0.y*d0.y + q0.z*d0.z + q0.w*d0.w;
        dp += q1.x*d1.x + q1.y*d1.y + q1.z*d1.z + q1.w*d1.w;
        ss += p0.x*p0.x + p0.y*p0.y + p0.z*p0.z + p0.w*p0.w;
        ss += p1.x*p1.x + p1.y*p1.y + p1.z*p1.z + p1.w*p1.w;
        ss += q0.x*q0.x + q0.y*q0.y + q0.z*q0.z + q0.w*q0.w;
        ss += q1.x*q1.x + q1.y*q1.y + q1.z*q1.z + q1.w*q1.w;
        v16h a;
        a[0]=(_Float16)p0.x;  a[1]=(_Float16)p0.y;  a[2]=(_Float16)p0.z;  a[3]=(_Float16)p0.w;
        a[4]=(_Float16)p1.x;  a[5]=(_Float16)p1.y;  a[6]=(_Float16)p1.z;  a[7]=(_Float16)p1.w;
        a[8]=(_Float16)q0.x;  a[9]=(_Float16)q0.y;  a[10]=(_Float16)q0.z; a[11]=(_Float16)q0.w;
        a[12]=(_Float16)q1.x; a[13]=(_Float16)q1.y; a[14]=(_Float16)q1.z; a[15]=(_Float16)q1.w;
        return a;
    };

    const int waveId = threadIdx.x >> 5;
    float* cw = sC[waveId];

    const int gwave   = blockIdx.x * WPB + waveId;
    const int nwaves  = gridDim.x * WPB;
    const int nchunks = N_NODES / KNEI;   // 3125 (exact)

    for (int ch = gwave; ch < nchunks; ch += nwaves) {
        const int nb = ch * KNEI;

        // prefetch first node's gathered neighbor rows
        {
            const long long im0 = nidx[(long long)nb * KNEI + mrow];
            const float* prow = feat + im0 * DFEAT + half * 64;
            __builtin_prefetch(prow, 0, 0);
            __builtin_prefetch(prow + 32, 0, 0);
        }

        // ================= Phase 1: batched ctr @ W2^T =================
        // A rows = the 16 nodes' own feature rows (row m -> node nb+m)
        const float* myc = feat + (long long)(nb + mrow) * DFEAT;
        float cn2 = 0.f;
        v16h at[4];
#pragma unroll
        for (int kc = 0; kc < 4; ++kc) at[kc] = buildA(myc, kc * 32, cn2);
        cn2 += __shfl_xor(cn2, 16, 32);          // lane pair covers 128 dims
        const float cnorm_all = sqrtf(cn2);      // ||ctr|| of node nb+mrow

        {
            v8f cacc[8] = {};
#pragma unroll
            for (int kc = 0; kc < 4; ++kc) {
#pragma unroll
                for (int t = 0; t < 8; ++t) {
                    v16h bm = loadB(t, 128 + kc * 32);   // W2 half
                    cacc[t] = __builtin_amdgcn_wmma_f32_16x16x32_f16(
                        false, at[kc], false, bm, (short)0, cacc[t], false, false);
                }
            }
            // park c[node][o] in this wave's LDS slab (in-order LDS per wave)
#pragma unroll
            for (int t = 0; t < 8; ++t)
#pragma unroll
                for (int r = 0; r < 8; ++r)
                    cw[(r + 8 * half) * DOUT + t * 16 + mrow] = cacc[t][r];
        }

        // ================= Phase 2: per-node neighbor pass =================
        for (int j = 0; j < KNEI; ++j) {
            const int n = nb + j;
            const float* crow = feat + (long long)n * DFEAT;
            const long long im = nidx[(long long)n * KNEI + mrow];
            const float* arow = feat + im * DFEAT;

            // prefetch next node's gathered rows
            if (j + 1 < KNEI) {
                const long long imn = nidx[(long long)(n + 1) * KNEI + mrow];
                const float* prow = feat + imn * DFEAT + half * 64;
                __builtin_prefetch(prow, 0, 0);
                __builtin_prefetch(prow + 32, 0, 0);
            }

            v8f acc[8] = {};
            float dotp = 0.f, nsq = 0.f;
#pragma unroll
            for (int kk = 0; kk < 4; ++kk) {
                v16h a = buildA_dot(arow, crow, kk * 32, dotp, nsq);
#pragma unroll
                for (int t = 0; t < 8; ++t) {
                    v16h bm = loadB(t, kk * 32);         // W1 half
                    acc[t] = __builtin_amdgcn_wmma_f32_16x16x32_f16(
                        false, a, false, bm, (short)0, acc[t], false, false);
                }
            }

            // cosine weight for neighbor m = lane&15
            dotp += __shfl_xor(dotp, 16, 32);
            nsq  += __shfl_xor(nsq,  16, 32);
            const float cnj = __shfl(cnorm_all, j, 32);  // ||ctr_n|| (uniform j)
            const float wgt = dotp / (sqrtf(nsq) * cnj);

            // redistribute weights to C-layout rows r + 8*half
            float wrw[8];
#pragma unroll
            for (int r = 0; r < 8; ++r) wrw[r] = __shfl(wgt, r + kb, 32);

            // this node's c row + bias, then weight & max over neighbors
#pragma unroll
            for (int t = 0; t < 8; ++t) {
                const float ct = cw[j * DOUT + t * 16 + mrow] + bv[t];
                float mx = (acc[t][0] + ct) * wrw[0];
#pragma unroll
                for (int r = 1; r < 8; ++r) {
                    float v = (acc[t][r] + ct) * wrw[r];
                    mx = fmaxf(mx, v);
                }
                mx = fmaxf(mx, __shfl_xor(mx, 16, 32));  // combine M halves
                if (lane < 16)
                    out[(long long)n * DOUT + t * 16 + mrow] = mx;
            }
        }
    }
}

extern "C" void kernel_launch(void* const* d_in, const int* in_sizes, int n_in,
                              void* d_out, int out_size, void* d_ws, size_t ws_size,
                              hipStream_t stream) {
    const float*     feat = (const float*)d_in[0];
    const long long* nidx = (const long long*)d_in[1];
    const float*     W    = (const float*)d_in[2];
    const float*     b    = (const float*)d_in[3];
    float*           out  = (float*)d_out;

    dim3 block(256);   // 8 waves per block (wave32)
    dim3 grid(391);    // 3128 waves ~= 3125 chunks of 16 nodes
    hipLaunchKernelGGL(neighconv_wmma, grid, block, 0, stream,
                       feat, nidx, W, b, out);
}
